// TransformerTextGeneratorWithLinguisticFeatures_28415503630368
// MI455X (gfx1250) — compile-verified, matrix operand order checked by reading, not run on hardware
//
#include <hip/hip_runtime.h>
#include <hip/hip_bf16.h>
#include <stdint.h>

// ---------------------------------------------------------------------------
// Model constants (from reference setup_inputs)
// ---------------------------------------------------------------------------
#define BB   4
#define SS   128
#define EE   512
#define HH   8
#define DH   64
#define LL   2
#define FFD  2048
#define VV   32000
#define MSTEPS 16
#define TT   (SS + MSTEPS)   // 144
#define NEGM (-1.0e9f)

typedef __attribute__((ext_vector_type(16))) __bf16 bf16x16;
typedef __attribute__((ext_vector_type(8)))  float  f32x8;
typedef unsigned short bfraw;                 // raw bf16 storage

union Frag {
    unsigned u[8];
    bf16x16  v;
};

__device__ __forceinline__ unsigned pack_bf16(float a, float b) {
    union { float f; unsigned u; } ua, ub;
    ua.f = a; ub.f = b;
    return (ub.u & 0xffff0000u) | (ua.u >> 16);
}
__device__ __forceinline__ bfraw bf_trunc(float a) {
    union { float f; unsigned u; } ua; ua.f = a;
    return (bfraw)(ua.u >> 16);
}

// Load one 16-bf16 WMMA fragment (A or B side) from bf16 row-major memory.
// Per CDNA5 ISA 7.12.2: lanes 0-15 hold K {kb..kb+7, 16+kb..16+kb+7}
// with kb = 0 (lanes 0-15) or 8 (lanes 16-31) -> two contiguous 16B chunks.
__device__ __forceinline__ void load_frag(Frag& f, const bfraw* base, int kb) {
    const uint4 lo = *(const uint4*)(base + kb);
    const uint4 hi = *(const uint4*)(base + kb + 16);
    f.u[0] = lo.x; f.u[1] = lo.y; f.u[2] = lo.z; f.u[3] = lo.w;
    f.u[4] = hi.x; f.u[5] = hi.y; f.u[6] = hi.z; f.u[7] = hi.w;
}

// ---------------------------------------------------------------------------
// fp32 -> bf16 bulk conversion (weights; once per launch). n % 8 == 0.
// ---------------------------------------------------------------------------
__global__ void cvt_kernel(const float* __restrict__ src,
                           bfraw* __restrict__ dst, size_t n) {
    const size_t i = ((size_t)blockIdx.x * blockDim.x + threadIdx.x) * 8;
    if (i >= n) return;
    const float4 a = *(const float4*)(src + i);
    const float4 b = *(const float4*)(src + i + 4);
    uint4 o;
    o.x = pack_bf16(a.x, a.y); o.y = pack_bf16(a.z, a.w);
    o.z = pack_bf16(b.x, b.y); o.w = pack_bf16(b.z, b.w);
    *(uint4*)(dst + i) = o;
}

// ---------------------------------------------------------------------------
// GEMM:  C[M,N] = A[M,K](bf16) * W[N,K](bf16)^T + bias, optional ReLU.
// One wave computes a 16(M) x 64(N) strip: one A fragment reused across four
// B fragments -> 4 independent back-to-back WMMAs per K-step.  Fragments are
// pure b128 loads (no packing VALU).  Writes fp32 C and a bf16 mirror.
// Requires M%16==0, N%64==0, K%32==0 (true for every GEMM here).
// ---------------------------------------------------------------------------
__global__ void gemm_bf16(const bfraw* __restrict__ A, int lda,
                          const bfraw* __restrict__ W, int ldw,
                          const float* __restrict__ bias,
                          float* __restrict__ Cf, bfraw* __restrict__ Cb,
                          int ldc, int Mr, int N, int K, int relu) {
    const int wave   = blockIdx.x * (blockDim.x >> 5) + (threadIdx.x >> 5);
    const int strips = N >> 6;
    const int tilesM = Mr >> 4;
    if (wave >= tilesM * strips) return;          // wave-uniform
    const int tm  = wave / strips, tn4 = wave % strips;
    const int l   = threadIdx.x & 31;
    const int mr  = tm * 16 + (l & 15);
    const int kb  = (l >> 4) * 8;

    f32x8 acc[4];
#pragma unroll
    for (int j = 0; j < 4; ++j) acc[j] = f32x8{0.f,0.f,0.f,0.f,0.f,0.f,0.f,0.f};

    const bfraw* Arow = A + (size_t)mr * lda;
    const bfraw* Wrow[4];
#pragma unroll
    for (int j = 0; j < 4; ++j)
        Wrow[j] = W + (size_t)(tn4 * 64 + j * 16 + (l & 15)) * ldw;

    for (int k0 = 0; k0 < K; k0 += 32) {
        if (k0 + 64 < K) {
            __builtin_prefetch(&Arow[k0 + 64], 0, 1);
            __builtin_prefetch(&Wrow[0][k0 + 64], 0, 1);
        }
        Frag a;
        load_frag(a, Arow + k0, kb);
#pragma unroll
        for (int j = 0; j < 4; ++j) {
            Frag b;
            load_frag(b, Wrow[j] + k0, kb);
            acc[j] = __builtin_amdgcn_wmma_f32_16x16x32_bf16(
                         false, a.v, false, b.v, (short)0, acc[j], false, false);
        }
    }
#pragma unroll
    for (int j = 0; j < 4; ++j) {
#pragma unroll
        for (int r = 0; r < 8; ++r) {
            const int m = tm * 16 + r + ((l >> 4) << 3);
            const int n = tn4 * 64 + j * 16 + (l & 15);
            float val = acc[j][r] + bias[n];
            if (relu) val = fmaxf(val, 0.f);
            Cf[(size_t)m * ldc + n] = val;
            Cb[(size_t)m * ldc + n] = bf_trunc(val);
        }
    }
}

// ---------------------------------------------------------------------------
// Fused attention over bf16 Q/K/V: one workgroup per (b, h, 16-row q-tile).
// scores via WMMA -> LDS fp32, masked softmax (half-wave shfl), P*V via WMMA.
// Output written bf16 (its only consumer is a GEMM A-operand).
// mode: 0 = none, 1 = causal (k > q masked), 2 = key-limit (k >= cur masked)
// ---------------------------------------------------------------------------
__global__ void attn_kernel(const bfraw* __restrict__ Q, int ldq,
                            const bfraw* __restrict__ Kp, int ldk,
                            const bfraw* __restrict__ Vp, int ldv,
                            bfraw* __restrict__ O,
                            int Lq, int Lk, int mode, int cur) {
    const float scale = 0.125f;                 // 1/sqrt(64)
    const int qt = blockIdx.x, h = blockIdx.y, b = blockIdx.z;
    const int wv = threadIdx.x >> 5, l = threadIdx.x & 31;
    const int kb = (l >> 4) * 8;

    __shared__ float sc[16][TT + 8];            // up to 16 x 144 scores (+pad)

    const bfraw* Qb = Q  + (size_t)b * Lq * ldq + h * DH;
    const bfraw* Kb = Kp + (size_t)b * Lk * ldk + h * DH;
    const bfraw* Vb = Vp + (size_t)b * Lk * ldv + h * DH;

    // ---- phase 1: scores (wave-uniform loop; EXEC full for WMMA) ----
    const int nkt = Lk >> 4;                    // Lk is 128 or 144 -> exact
    for (int kt = wv; kt < nkt; kt += 8) {
        f32x8 acc = {0.f, 0.f, 0.f, 0.f, 0.f, 0.f, 0.f, 0.f};
        const bfraw* Qr = Qb + (size_t)(qt * 16 + (l & 15)) * ldq;
        const bfraw* Kr = Kb + (size_t)(kt * 16 + (l & 15)) * ldk;
#pragma unroll
        for (int kk = 0; kk < 2; ++kk) {        // Dh = 64 = 2 * 32
            Frag a, bf;
            load_frag(a,  Qr + kk * 32, kb);
            load_frag(bf, Kr + kk * 32, kb);
            acc = __builtin_amdgcn_wmma_f32_16x16x32_bf16(
                      false, a.v, false, bf.v, (short)0, acc, false, false);
        }
#pragma unroll
        for (int r = 0; r < 8; ++r) {
            const int ml = r + ((l >> 4) << 3);
            const int nl = l & 15;
            const int qa = qt * 16 + ml, ka = kt * 16 + nl;
            float s = acc[r] * scale;
            if (mode == 1 && ka > qa)  s = NEGM;
            if (mode == 2 && ka >= cur) s = NEGM;
            sc[ml][ka] = s;
        }
    }
    __syncthreads();

    // ---- phase 2: row softmax, 16 threads per row ----
    {
        const int row = threadIdx.x >> 4, li = threadIdx.x & 15;
        float mx = -3.4e38f;
        for (int k = li; k < Lk; k += 16) mx = fmaxf(mx, sc[row][k]);
#pragma unroll
        for (int m = 8; m >= 1; m >>= 1) mx = fmaxf(mx, __shfl_xor(mx, m, 16));
        float sum = 0.f;
        for (int k = li; k < Lk; k += 16) {
            const float e = __expf(sc[row][k] - mx);
            sc[row][k] = e;
            sum += e;
        }
#pragma unroll
        for (int m = 8; m >= 1; m >>= 1) sum += __shfl_xor(sum, m, 16);
        const float inv = 1.f / sum;
        for (int k = li; k < Lk; k += 16) sc[row][k] *= inv;
    }
    __syncthreads();

    // ---- phase 3: O = P * V ; waves 0..3 each do one 16-wide slice of Dh ----
    if (wv < 4) {
        const int nt = wv;
        f32x8 acc = {0.f, 0.f, 0.f, 0.f, 0.f, 0.f, 0.f, 0.f};
        const int mrow = l & 15;
        const int ncol = nt * 16 + (l & 15);
        for (int k0 = 0; k0 < Lk; k0 += 32) {   // Lk=144: remainder zero-padded
            Frag a, bf;
#pragma unroll
            for (int v = 0; v < 8; ++v) {
                const int k = k0 + ((v < 4) ? (kb + 2 * v) : (16 + kb + 2 * (v - 4)));
                // Lk even, k even -> pair is in/out of range together
                const float2 p2 = (k < Lk) ? *(const float2*)(&sc[mrow][k])
                                           : float2{0.f, 0.f};
                a.u[v] = pack_bf16(p2.x, p2.y);
                const int kc = (k < Lk) ? k : 0;   // P==0 there, value irrelevant
                const unsigned v0 = Vb[(size_t)kc * ldv + ncol];
                const unsigned v1 = Vb[(size_t)(kc + 1) * ldv + ncol];
                bf.u[v] = (v1 << 16) | v0;
            }
            acc = __builtin_amdgcn_wmma_f32_16x16x32_bf16(
                      false, a.v, false, bf.v, (short)0, acc, false, false);
        }
#pragma unroll
        for (int r = 0; r < 8; ++r) {
            const int m = r + ((l >> 4) << 3);
            const int q = qt * 16 + m;
            O[((size_t)b * Lq + q) * EE + h * DH + nt * 16 + (l & 15)] =
                bf_trunc(acc[r]);
        }
    }
}

// ---------------------------------------------------------------------------
// Fused residual + LayerNorm over E=512; writes fp32 + bf16 mirror.
// ---------------------------------------------------------------------------
__global__ void ln_kernel(const float* __restrict__ X, const float* __restrict__ R,
                          const float* __restrict__ g, const float* __restrict__ bta,
                          float* __restrict__ out, bfraw* __restrict__ outb) {
    const int row = blockIdx.x, t = threadIdx.x;
    const size_t base = (size_t)row * EE;
    float v0 = X[base + t], v1 = X[base + t + 256];
    if (R) { v0 += R[base + t]; v1 += R[base + t + 256]; }
    __shared__ float red[8];
    float s = v0 + v1;
#pragma unroll
    for (int m = 16; m >= 1; m >>= 1) s += __shfl_xor(s, m, 32);
    if ((t & 31) == 0) red[t >> 5] = s;
    __syncthreads();
    float mean = 0.f;
#pragma unroll
    for (int i = 0; i < 8; ++i) mean += red[i];
    mean *= (1.f / EE);
    __syncthreads();
    const float d0 = v0 - mean, d1 = v1 - mean;
    float s2 = d0 * d0 + d1 * d1;
#pragma unroll
    for (int m = 16; m >= 1; m >>= 1) s2 += __shfl_xor(s2, m, 32);
    if ((t & 31) == 0) red[t >> 5] = s2;
    __syncthreads();
    float var = 0.f;
#pragma unroll
    for (int i = 0; i < 8; ++i) var += red[i];
    var *= (1.f / EE);
    const float inv = rsqrtf(var + 1e-5f);
    const float o0 = d0 * inv * g[t] + bta[t];
    const float o1 = d1 * inv * g[t + 256] + bta[t + 256];
    out[base + t] = o0;          out[base + t + 256] = o1;
    outb[base + t] = bf_trunc(o0); outb[base + t + 256] = bf_trunc(o1);
}

// ---------------------------------------------------------------------------
// Utility kernels
// ---------------------------------------------------------------------------
__global__ void gather_kernel(const int* __restrict__ idx,
                              const float* __restrict__ emb,
                              const float* __restrict__ pos,
                              float* __restrict__ out, bfraw* __restrict__ outb) {
    const int row = blockIdx.x;                 // b*S + s
    const int tok = idx[row];
    const int s = row % SS;
    for (int e = threadIdx.x; e < EE; e += blockDim.x) {
        float v = emb[(size_t)tok * EE + e];
        if (pos) v += pos[(size_t)s * EE + e];
        out[(size_t)row * EE + e] = v;
        outb[(size_t)row * EE + e] = bf_trunc(v);
    }
}

__global__ void bufinit_kernel(const float* __restrict__ x,
                               float* __restrict__ buf, bfraw* __restrict__ bufb) {
    const size_t i = (size_t)blockIdx.x * blockDim.x + threadIdx.x;
    const size_t total = (size_t)BB * TT * EE;
    if (i >= total) return;
    const int e = (int)(i % EE);
    const size_t r = i / EE;
    const int tpos = (int)(r % TT);
    const int b = (int)(r / TT);
    const float v = (tpos < SS) ? x[((size_t)b * SS + tpos) * EE + e] : 0.f;
    buf[i] = v;
    bufb[i] = bf_trunc(v);
}

__global__ void lastrow_kernel(const float* __restrict__ tgt,
                               bfraw* __restrict__ last16b, int cur) {
    const int row = blockIdx.x;                 // 0..15, rows >= B are zero pad
    for (int e = threadIdx.x; e < EE; e += blockDim.x) {
        float v = 0.f;
        if (row < BB) v = tgt[((size_t)row * TT + (cur - 1)) * EE + e];
        last16b[(size_t)row * EE + e] = bf_trunc(v);
    }
}

__global__ void argmax_embed_kernel(const float* __restrict__ logits,
                                    const float* __restrict__ emb,
                                    float* __restrict__ buf,
                                    bfraw* __restrict__ bufb,
                                    float* __restrict__ outtok,
                                    int cur, int step) {
    const int b = blockIdx.x, t = threadIdx.x;
    __shared__ float bm[256];
    __shared__ int   bi[256];
    float best = -3.4e38f; int besti = 0;
    for (int v = t; v < VV; v += 256) {
        const float lv = logits[(size_t)b * VV + v];
        if (lv > best) { best = lv; besti = v; }
    }
    bm[t] = best; bi[t] = besti;
    __syncthreads();
    for (int s = 128; s >= 1; s >>= 1) {
        if (t < s) {
            if (bm[t + s] > bm[t] ||
                (bm[t + s] == bm[t] && bi[t + s] < bi[t])) {
                bm[t] = bm[t + s]; bi[t] = bi[t + s];
            }
        }
        __syncthreads();
    }
    const int tok = bi[0];
    if (t == 0) outtok[(size_t)b * MSTEPS + step] = (float)tok;
    for (int e = t; e < EE; e += 256) {
        const float v = emb[(size_t)tok * EE + e];
        buf[((size_t)b * TT + cur) * EE + e] = v;
        bufb[((size_t)b * TT + cur) * EE + e] = bf_trunc(v);
    }
}

// ---------------------------------------------------------------------------
// Host-side orchestration
// ---------------------------------------------------------------------------
static inline void launch_gemm(const bfraw* A, int lda, const bfraw* W, int ldw,
                               const float* bias, float* Cf, bfraw* Cb, int ldc,
                               int Mr, int N, int K, int relu, hipStream_t s) {
    const int tiles = (Mr / 16) * (N / 64);
    const int blocks = (tiles + 7) / 8;
    gemm_bf16<<<blocks, 256, 0, s>>>(A, lda, W, ldw, bias, Cf, Cb, ldc, Mr, N, K, relu);
}

static inline void cvt(const float* s, bfraw* d, size_t n, hipStream_t st) {
    const size_t thr = (n + 7) / 8;
    cvt_kernel<<<(unsigned)((thr + 255) / 256), 256, 0, st>>>(s, d, n);
}

extern "C" void kernel_launch(void* const* d_in, const int* in_sizes, int n_in,
                              void* d_out, int out_size, void* d_ws, size_t ws_size,
                              hipStream_t stream) {
    (void)in_sizes; (void)n_in; (void)out_size; (void)ws_size;
    const int* input_seq = (const int*)d_in[0];
    const int* pos_tags  = (const int*)d_in[1];
    const int* ner_tags  = (const int*)d_in[2];
    const float* embedding = (const float*)d_in[4];
    const float* pos_emb   = (const float*)d_in[5];
    const float* fc_pos_w  = (const float*)d_in[6];
    const float* fc_pos_b  = (const float*)d_in[7];
    const float* fc_ner_w  = (const float*)d_in[8];
    const float* fc_ner_b  = (const float*)d_in[9];
    const float* ln_g = (const float*)d_in[10];
    const float* ln_b = (const float*)d_in[11];
    const float* asi_w = (const float*)d_in[12]; const float* asi_b = (const float*)d_in[13];
    const float* aso_w = (const float*)d_in[14]; const float* aso_b = (const float*)d_in[15];
    const float* api_w = (const float*)d_in[16]; const float* api_b = (const float*)d_in[17];
    const float* apo_w = (const float*)d_in[18]; const float* apo_b = (const float*)d_in[19];
    const float* ani_w = (const float*)d_in[20]; const float* ani_b = (const float*)d_in[21];
    const float* ano_w = (const float*)d_in[22]; const float* ano_b = (const float*)d_in[23];
    const float* dsi_w = (const float*)d_in[24]; const float* dsi_b = (const float*)d_in[25];
    const float* dso_w = (const float*)d_in[26]; const float* dso_b = (const float*)d_in[27];
    const float* dci_w = (const float*)d_in[28]; const float* dci_b = (const float*)d_in[29];
    const float* dco_w = (const float*)d_in[30]; const float* dco_b = (const float*)d_in[31];
    const float* d1_w  = (const float*)d_in[32]; const float* d1_b  = (const float*)d_in[33];
    const float* d2_w  = (const float*)d_in[34]; const float* d2_b  = (const float*)d_in[35];
    const float* dln_g = (const float*)d_in[36]; const float* dln_b = (const float*)d_in[37];
    const float* fo_w  = (const float*)d_in[38]; const float* fo_b  = (const float*)d_in[39];

    // ---------------- workspace layout ----------------
    float* p = (float*)d_ws;
    float* gbuf   = p; p += (size_t)SS * BB * EE;
    float* srcb   = p; p += (size_t)SS * BB * EE;
    float* xbuf   = p; p += (size_t)SS * BB * EE;
    float* x2buf  = p; p += (size_t)SS * BB * EE;
    float* pfb    = p; p += (size_t)SS * BB * EE;
    float* qkv    = p; p += (size_t)BB * TT * 3 * EE;
    float* qb     = p; p += (size_t)BB * TT * EE;
    float* kvb    = p; p += (size_t)BB * TT * 2 * EE;
    float* proj   = p; p += (size_t)BB * TT * EE;
    float* bufB   = p; p += (size_t)BB * TT * EE;
    float* tgtb   = p; p += (size_t)BB * TT * EE;
    float* ffh    = p; p += (size_t)BB * TT * FFD;
    float* logits = p; p += (size_t)16 * VV;

    bfraw* q2 = (bfraw*)p;
    auto ab = [&](size_t n) { bfraw* r = q2; q2 += n; return r; };
    // bf16 activation mirrors
    bfraw* gbufb   = ab((size_t)SS * BB * EE);
    bfraw* srcbb   = ab((size_t)SS * BB * EE);
    bfraw* xbufb   = ab((size_t)SS * BB * EE);
    bfraw* x2bufb  = ab((size_t)SS * BB * EE);
    bfraw* pfbb    = ab((size_t)SS * BB * EE);
    bfraw* qkvb    = ab((size_t)BB * TT * 3 * EE);
    bfraw* qbb     = ab((size_t)BB * TT * EE);
    bfraw* kvbb    = ab((size_t)BB * TT * 2 * EE);
    bfraw* attnob  = ab((size_t)BB * TT * EE);
    bfraw* projb   = ab((size_t)BB * TT * EE);
    bfraw* bufBb   = ab((size_t)BB * TT * EE);
    bfraw* tgtbb   = ab((size_t)BB * TT * EE);
    bfraw* ffhb    = ab((size_t)BB * TT * FFD);
    bfraw* logitsb = ab((size_t)16 * VV);
    bfraw* last16b = ab((size_t)16 * EE);
    // bf16 weights
    bfraw* wasi = ab((size_t)3 * EE * EE);
    bfraw* waso = ab((size_t)EE * EE);
    bfraw* wapi = ab((size_t)3 * EE * EE);
    bfraw* wapo = ab((size_t)EE * EE);
    bfraw* wani = ab((size_t)3 * EE * EE);
    bfraw* wano = ab((size_t)EE * EE);
    bfraw* wfcp = ab((size_t)EE * EE);
    bfraw* wfcn = ab((size_t)EE * EE);
    bfraw* wdsi = ab((size_t)LL * 3 * EE * EE);
    bfraw* wdso = ab((size_t)LL * EE * EE);
    bfraw* wdci = ab((size_t)LL * 3 * EE * EE);
    bfraw* wdco = ab((size_t)LL * EE * EE);
    bfraw* wd1  = ab((size_t)LL * FFD * EE);
    bfraw* wd2  = ab((size_t)LL * EE * FFD);
    bfraw* wfo  = ab((size_t)VV * EE);

    const int encRows = BB * SS;   // 512
    const int decRows = BB * TT;   // 576

    // ---------------- weight conversion (once per launch) ----------------
    cvt(asi_w, wasi, (size_t)3 * EE * EE, stream);
    cvt(aso_w, waso, (size_t)EE * EE, stream);
    cvt(api_w, wapi, (size_t)3 * EE * EE, stream);
    cvt(apo_w, wapo, (size_t)EE * EE, stream);
    cvt(ani_w, wani, (size_t)3 * EE * EE, stream);
    cvt(ano_w, wano, (size_t)EE * EE, stream);
    cvt(fc_pos_w, wfcp, (size_t)EE * EE, stream);
    cvt(fc_ner_w, wfcn, (size_t)EE * EE, stream);
    cvt(dsi_w, wdsi, (size_t)LL * 3 * EE * EE, stream);
    cvt(dso_w, wdso, (size_t)LL * EE * EE, stream);
    cvt(dci_w, wdci, (size_t)LL * 3 * EE * EE, stream);
    cvt(dco_w, wdco, (size_t)LL * EE * EE, stream);
    cvt(d1_w, wd1, (size_t)LL * FFD * EE, stream);
    cvt(d2_w, wd2, (size_t)LL * EE * FFD, stream);
    cvt(fo_w, wfo, (size_t)VV * EE, stream);

    // ===================== encoder =====================
    gather_kernel<<<encRows, 256, 0, stream>>>(input_seq, embedding, pos_emb, gbuf, gbufb);
    ln_kernel<<<encRows, 256, 0, stream>>>(gbuf, nullptr, ln_g, ln_b, srcb, srcbb);

    // self-attention block
    launch_gemm(srcbb, EE, wasi, EE, asi_b, qkv, qkvb, 3 * EE, encRows, 3 * EE, EE, 0, stream);
    attn_kernel<<<dim3(SS / 16, HH, BB), 256, 0, stream>>>(
        qkvb, 3 * EE, qkvb + EE, 3 * EE, qkvb + 2 * EE, 3 * EE, attnob, SS, SS, 0, 0);
    launch_gemm(attnob, EE, waso, EE, aso_b, xbuf, xbufb, EE, encRows, EE, EE, 0, stream);

    // cross vs POS features
    gather_kernel<<<encRows, 256, 0, stream>>>(pos_tags, embedding, nullptr, gbuf, gbufb);
    launch_gemm(gbufb, EE, wfcp, EE, fc_pos_b, pfb, pfbb, EE, encRows, EE, EE, 0, stream);
    launch_gemm(xbufb, EE, wapi, EE, api_b, qb, qbb, EE, encRows, EE, EE, 0, stream);
    launch_gemm(pfbb, EE, wapi + (size_t)EE * EE, EE, api_b + EE, kvb, kvbb, 2 * EE,
                encRows, 2 * EE, EE, 0, stream);
    attn_kernel<<<dim3(SS / 16, HH, BB), 256, 0, stream>>>(
        qbb, EE, kvbb, 2 * EE, kvbb + EE, 2 * EE, attnob, SS, SS, 0, 0);
    launch_gemm(attnob, EE, wapo, EE, apo_b, x2buf, x2bufb, EE, encRows, EE, EE, 0, stream);

    // cross vs NER features
    gather_kernel<<<encRows, 256, 0, stream>>>(ner_tags, embedding, nullptr, gbuf, gbufb);
    launch_gemm(gbufb, EE, wfcn, EE, fc_ner_b, pfb, pfbb, EE, encRows, EE, EE, 0, stream);
    launch_gemm(x2bufb, EE, wani, EE, ani_b, qb, qbb, EE, encRows, EE, EE, 0, stream);
    launch_gemm(pfbb, EE, wani + (size_t)EE * EE, EE, ani_b + EE, kvb, kvbb, 2 * EE,
                encRows, 2 * EE, EE, 0, stream);
    attn_kernel<<<dim3(SS / 16, HH, BB), 256, 0, stream>>>(
        qbb, EE, kvbb, 2 * EE, kvbb + EE, 2 * EE, attnob, SS, SS, 0, 0);
    launch_gemm(attnob, EE, wano, EE, ano_b, xbuf, xbufb, EE, encRows, EE, EE, 0, stream);

    // init generation buffer (fp32 + bf16)
    {
        const size_t total = (size_t)BB * TT * EE;
        bufinit_kernel<<<(unsigned)((total + 255) / 256), 256, 0, stream>>>(xbuf, bufB, bufBb);
    }

    // ===================== decode loop =====================
    for (int i = 0; i < MSTEPS; ++i) {
        const int cur = SS + i;
        hipMemcpyAsync(tgtb, bufB, sizeof(float) * (size_t)BB * TT * EE,
                       hipMemcpyDeviceToDevice, stream);
        hipMemcpyAsync(tgtbb, bufBb, sizeof(bfraw) * (size_t)BB * TT * EE,
                       hipMemcpyDeviceToDevice, stream);
        for (int l = 0; l < LL; ++l) {
            const bfraw* wSI = wdsi + (size_t)l * 3 * EE * EE;
            const float* bSI = dsi_b + (size_t)l * 3 * EE;
            const bfraw* wSO = wdso + (size_t)l * EE * EE;
            const float* bSO = dso_b + (size_t)l * EE;
            const bfraw* wCI = wdci + (size_t)l * 3 * EE * EE;
            const float* bCI = dci_b + (size_t)l * 3 * EE;
            const bfraw* wCO = wdco + (size_t)l * EE * EE;
            const float* bCO = dco_b + (size_t)l * EE;
            const bfraw* w1  = wd1  + (size_t)l * FFD * EE;
            const float* b1  = d1_b + (size_t)l * FFD;
            const bfraw* w2  = wd2  + (size_t)l * EE * FFD;
            const float* b2  = d2_b + (size_t)l * EE;
            const float* lg  = dln_g + (size_t)l * 3 * EE;
            const float* lb  = dln_b + (size_t)l * 3 * EE;

            // self-attention (causal)
            launch_gemm(tgtbb, EE, wSI, EE, bSI, qkv, qkvb, 3 * EE, decRows, 3 * EE, EE, 0, stream);
            attn_kernel<<<dim3(TT / 16, HH, BB), 256, 0, stream>>>(
                qkvb, 3 * EE, qkvb + EE, 3 * EE, qkvb + 2 * EE, 3 * EE, attnob, TT, TT, 1, 0);
            launch_gemm(attnob, EE, wSO, EE, bSO, proj, projb, EE, decRows, EE, EE, 0, stream);
            ln_kernel<<<decRows, 256, 0, stream>>>(tgtb, proj, lg, lb, tgtb, tgtbb);

            // cross-attention vs memory (key-limit mask)
            launch_gemm(tgtbb, EE, wCI, EE, bCI, qb, qbb, EE, decRows, EE, EE, 0, stream);
            launch_gemm(bufBb, EE, wCI + (size_t)EE * EE, EE, bCI + EE, kvb, kvbb, 2 * EE,
                        decRows, 2 * EE, EE, 0, stream);
            attn_kernel<<<dim3(TT / 16, HH, BB), 256, 0, stream>>>(
                qbb, EE, kvbb, 2 * EE, kvbb + EE, 2 * EE, attnob, TT, TT, 2, cur);
            launch_gemm(attnob, EE, wCO, EE, bCO, proj, projb, EE, decRows, EE, EE, 0, stream);
            ln_kernel<<<decRows, 256, 0, stream>>>(tgtb, proj, lg + EE, lb + EE, tgtb, tgtbb);

            // FFN (relu)
            launch_gemm(tgtbb, EE, w1, EE, b1, ffh, ffhb, FFD, decRows, FFD, EE, 1, stream);
            launch_gemm(ffhb, FFD, w2, FFD, b2, proj, projb, EE, decRows, EE, FFD, 0, stream);
            ln_kernel<<<decRows, 256, 0, stream>>>(tgtb, proj, lg + 2 * EE, lb + 2 * EE, tgtb, tgtbb);
        }
        // logits for last valid row, argmax, append embedding
        lastrow_kernel<<<16, 256, 0, stream>>>(tgtb, last16b, cur);
        launch_gemm(last16b, EE, wfo, EE, fo_b, logits, logitsb, VV, 16, VV, EE, 0, stream);
        argmax_embed_kernel<<<BB, 256, 0, stream>>>(logits, embedding, bufB, bufBb,
                                                    (float*)d_out, cur, i);
    }
}
